// GambaSP_7653631722029
// MI455X (gfx1250) — matrix operand order repaired and checked by reading, not compiled, not run on hardware
//
#include <hip/hip_runtime.h>
#include <hip/hip_bf16.h>
#include <math.h>

// ---------------- types / constants ----------------
typedef _Float16 f16_t;
typedef __attribute__((ext_vector_type(16))) _Float16 v16h;
typedef __attribute__((ext_vector_type(8)))  float    v8f;

#define NNODES  65536
#define NGRAPHS 64
#define MNODES  1024
#define NEDGES  1048576
#define IN_DIM  32
#define HDIM    64
#define OUT_DIM 10
#define IDIM    128
#define NSTATE  16
#define NRANK   4
#define EPSV    1e-5f

// ---------------- WMMA helpers (wave32; A and B share a contiguous k order) -------
// A: 16x32 f16. lanes 0-15: M=0..15, K = 0..15; lanes 16-31: K = 16..31
// Two contiguous 16B LDS reads per fragment.
__device__ __forceinline__ v16h wmma_load_A(const f16_t* sA, int lda, int k0, int lane) {
  int row  = lane & 15;
  int koff = (lane >> 4) << 4;           // 0 or 16
  const f16_t* p = sA + row * lda + k0 + koff;
  v16h a;
#pragma unroll
  for (int i = 0; i < 8; ++i) a[i] = p[i];
#pragma unroll
  for (int i = 0; i < 8; ++i) a[8 + i] = p[8 + i];
  return a;
}
// B stored TRANSPOSED in LDS as (N x K) rows with row stride ldk (padded).
// lane col = n0 + lane&15; lanes 0-15: K=k0..k0+15, lanes 16-31: K=k0+16..k0+31.
// One contiguous 32B read per fragment (2x b128).
__device__ __forceinline__ v16h wmma_load_Bt(const f16_t* sBt, int ldk, int k0, int n0, int lane) {
  int col  = n0 + (lane & 15);
  int koff = (lane >> 4) << 4;           // 0 or 16
  const f16_t* p = sBt + col * ldk + k0 + koff;
  v16h b;
#pragma unroll
  for (int i = 0; i < 16; ++i) b[i] = p[i];
  return b;
}
__device__ __forceinline__ v8f wmma_step(v8f acc, v16h a, v16h b) {
  return __builtin_amdgcn_wmma_f32_16x16x32_f16(false, a, false, b, (short)0, acc,
                                                false, false);
}
// C/D map: VGPR r -> row = r + 8*(lane>=16), col = lane&15 (+ wave's n0)

__device__ __forceinline__ float sigmoid_f(float x) { return 1.f / (1.f + __expf(-x)); }
__device__ __forceinline__ float silu_f(float x)    { return x / (1.f + __expf(-x)); }

// LayerNorm over 64 features of one row in LDS (f32) -> ReLU -> f16 tile.
__device__ __forceinline__ void row_ln_relu_f16(const float* sC, f16_t* sA, int lda,
                                                const float* g, const float* b, int row) {
  float m = 0.f;
  for (int c = 0; c < HDIM; ++c) m += sC[row * HDIM + c];
  m *= (1.f / HDIM);
  float v = 0.f;
  for (int c = 0; c < HDIM; ++c) { float d = sC[row * HDIM + c] - m; v += d * d; }
  float rs = rsqrtf(v * (1.f / HDIM) + EPSV);
  for (int c = 0; c < HDIM; ++c) {
    float y = (sC[row * HDIM + c] - m) * rs * g[c] + b[c];
    sA[row * lda + c] = (f16_t)fmaxf(y, 0.f);
  }
}

// ---------------- f32 (K x O) -> f16 transposed (O x K) ----------------
__global__ void k_cast_f16_t(const float* __restrict__ s, f16_t* __restrict__ d,
                             int K, int O) {
  int i = blockIdx.x * blockDim.x + threadIdx.x;   // linear index into dst (O x K)
  if (i < K * O) {
    int o = i / K, k = i - o * K;
    d[i] = (f16_t)s[(size_t)k * O + o];
  }
}

// ---------------- encoder: 32 ->64(LN,ReLU) ->64(LN,ReLU) ->64 ----------------
struct EncArgs {
  const float* x;
  const f16_t *W0t, *W1t, *W2t;           // f16 weights, transposed (out x in)
  const float *b0, *ln0g, *ln0b;
  const float *b1, *ln1g, *ln1b;
  const float *bo;
  float* h; f16_t* h16;
};
__global__ __launch_bounds__(128) void k_encoder(EncArgs A) {
  __shared__ f16_t sA[16 * 72];
  __shared__ f16_t sW[64 * 72];            // transposed weight panel, ldk = 72
  __shared__ float sC[16 * 64];
  const int tid = threadIdx.x, wave = tid >> 5, lane = tid & 31;
  const int n0 = blockIdx.x * 16;
  const int hi = (lane >> 4), col = wave * 16 + (lane & 15);

  // stage x tile (16x32) as f16
  for (int i = tid; i < 16 * IN_DIM; i += 128) {
    int r = i >> 5, c = i & 31;
    sA[r * 72 + c] = (f16_t)A.x[(size_t)(n0 + r) * IN_DIM + c];
  }
  for (int i = tid; i < HDIM * IN_DIM; i += 128) {   // W0t: 64 x 32
    int o = i >> 5, k = i & 31;
    sW[o * 72 + k] = A.W0t[i];
  }
  __syncthreads();

  v8f acc = {};
  acc = wmma_step(acc, wmma_load_A(sA, 72, 0, lane), wmma_load_Bt(sW, 72, 0, wave * 16, lane));
#pragma unroll
  for (int r = 0; r < 8; ++r) sC[(r + (hi << 3)) * HDIM + col] = acc[r] + A.b0[col];
  __syncthreads();
  if (tid < 16) row_ln_relu_f16(sC, sA, 72, A.ln0g, A.ln0b, tid);
  for (int i = tid; i < HDIM * HDIM; i += 128) {     // W1t: 64 x 64
    int o = i >> 6, k = i & 63;
    sW[o * 72 + k] = A.W1t[i];
  }
  __syncthreads();

  acc = (v8f){};
#pragma unroll
  for (int k0 = 0; k0 < HDIM; k0 += 32)
    acc = wmma_step(acc, wmma_load_A(sA, 72, k0, lane), wmma_load_Bt(sW, 72, k0, wave * 16, lane));
#pragma unroll
  for (int r = 0; r < 8; ++r) sC[(r + (hi << 3)) * HDIM + col] = acc[r] + A.b1[col];
  __syncthreads();
  if (tid < 16) row_ln_relu_f16(sC, sA, 72, A.ln1g, A.ln1b, tid);
  for (int i = tid; i < HDIM * HDIM; i += 128) {
    int o = i >> 6, k = i & 63;
    sW[o * 72 + k] = A.W2t[i];
  }
  __syncthreads();

  acc = (v8f){};
#pragma unroll
  for (int k0 = 0; k0 < HDIM; k0 += 32)
    acc = wmma_step(acc, wmma_load_A(sA, 72, k0, lane), wmma_load_Bt(sW, 72, k0, wave * 16, lane));
#pragma unroll
  for (int r = 0; r < 8; ++r) {
    int row = r + (hi << 3);
    float v = acc[r] + A.bo[col];
    A.h[(size_t)(n0 + row) * HDIM + col]   = v;
    A.h16[(size_t)(n0 + row) * HDIM + col] = (f16_t)v;
  }
}

// ---------------- edge conv: per-edge MLP + gated scatter ----------------
struct ConvArgs {
  const int* src; const int* dst;
  const float* edge_attr;                  // NEDGES x 2
  const f16_t* h16;                        // NNODES x 64
  const f16_t *W1t, *W2t;                  // transposed: 64x128, 64x64 (f16)
  const float *b1, *ln1g, *ln1b, *b2;
  const float *sob_w, *sob_b, *bnd_w, *bnd_b;
  float* aggr;                             // NNODES x 64 (f32, zeroed)
};
__global__ __launch_bounds__(128) void k_edge_conv(ConvArgs A) {
  __shared__ int   sSrc[16], sDst[16];
  __shared__ f16_t sA[16 * 136];
  __shared__ f16_t sW1[HDIM * 136];        // 64 x 128 transposed, ldk = 136
  __shared__ f16_t sW2[HDIM * 72];         // 64 x 64 transposed, ldk = 72
  __shared__ float sC[16 * 64];
  __shared__ float sG[16 * 64];
  const int tid = threadIdx.x, wave = tid >> 5, lane = tid & 31;
  const int e0 = blockIdx.x * 16;
  const int hi = (lane >> 4), col = wave * 16 + (lane & 15);

  if (tid < 16) { sSrc[tid] = A.src[e0 + tid]; sDst[tid] = A.dst[e0 + tid]; }
  if (tid == 0) { __builtin_prefetch(&A.src[e0 + 16], 0, 1); __builtin_prefetch(&A.dst[e0 + 16], 0, 1); }
  __syncthreads();

  // gather concat [x_dst | x_src] as 16x128 f16
  for (int i = tid; i < 16 * IDIM; i += 128) {
    int r = i >> 7, c = i & 127;
    int node = (c < HDIM) ? sDst[r] : sSrc[r];
    sA[r * 136 + c] = A.h16[(size_t)node * HDIM + (c & 63)];
  }
  for (int i = tid; i < HDIM * IDIM; i += 128) {     // W1t: 64 x 128
    int o = i >> 7, k = i & 127;
    sW1[o * 136 + k] = A.W1t[i];
  }
  for (int i = tid; i < HDIM * HDIM; i += 128) {     // W2t: 64 x 64
    int o = i >> 6, k = i & 63;
    sW2[o * 72 + k] = A.W2t[i];
  }
  // per-edge gate = sigmoid(ea0*sobel_w + sobel_b + ea1*bound_w + bound_b)
  for (int i = tid; i < 16 * HDIM; i += 128) {
    int r = i >> 6, c = i & 63;
    float ea0 = A.edge_attr[(size_t)(e0 + r) * 2 + 0];
    float ea1 = A.edge_attr[(size_t)(e0 + r) * 2 + 1];
    sG[i] = sigmoid_f(ea0 * A.sob_w[c] + A.sob_b[c] + ea1 * A.bnd_w[c] + A.bnd_b[c]);
  }
  __syncthreads();

  v8f acc = {};
#pragma unroll
  for (int k0 = 0; k0 < IDIM; k0 += 32)
    acc = wmma_step(acc, wmma_load_A(sA, 136, k0, lane), wmma_load_Bt(sW1, 136, k0, wave * 16, lane));
#pragma unroll
  for (int r = 0; r < 8; ++r) sC[(r + (hi << 3)) * HDIM + col] = acc[r] + A.b1[col];
  __syncthreads();
  if (tid < 16) row_ln_relu_f16(sC, sA, 136, A.ln1g, A.ln1b, tid);
  __syncthreads();

  acc = (v8f){};
#pragma unroll
  for (int k0 = 0; k0 < HDIM; k0 += 32)
    acc = wmma_step(acc, wmma_load_A(sA, 136, k0, lane), wmma_load_Bt(sW2, 72, k0, wave * 16, lane));
#pragma unroll
  for (int r = 0; r < 8; ++r) {
    int row = r + (hi << 3);
    float v = (acc[r] + A.b2[col]) * sG[row * HDIM + col];
    atomicAdd(&A.aggr[(size_t)sDst[row] * HDIM + col], v);  // segment_sum over dst
  }
}

// ---------------- xs = LN(h + aggr) -> f16 (warp per row, wave32) ----------------
__global__ __launch_bounds__(256) void k_fn_ln(const float* __restrict__ h,
                                               const float* __restrict__ aggr,
                                               const float* __restrict__ g,
                                               const float* __restrict__ b,
                                               f16_t* __restrict__ xs16) {
  int warp = threadIdx.x >> 5, lane = threadIdx.x & 31;
  size_t node = (size_t)blockIdx.x * 8 + warp;
  float v0 = h[node * HDIM + lane]      + aggr[node * HDIM + lane];
  float v1 = h[node * HDIM + lane + 32] + aggr[node * HDIM + lane + 32];
  float s = v0 + v1;
#pragma unroll
  for (int m = 16; m >= 1; m >>= 1) s += __shfl_xor(s, m, 32);
  float mean = s * (1.f / HDIM);
  float d0 = v0 - mean, d1 = v1 - mean;
  float q = d0 * d0 + d1 * d1;
#pragma unroll
  for (int m = 16; m >= 1; m >>= 1) q += __shfl_xor(q, m, 32);
  float rs = rsqrtf(q * (1.f / HDIM) + EPSV);
  xs16[node * HDIM + lane]      = (f16_t)(d0 * rs * g[lane] + b[lane]);
  xs16[node * HDIM + lane + 32] = (f16_t)(d1 * rs * g[lane + 32] + b[lane + 32]);
}

// ---------------- alpha = (xd @ theta)^T ; ax = alpha @ xd  (per graph) ----------------
__global__ __launch_bounds__(256) void k_alpha_ax(const float* __restrict__ h,
                                                  const float* __restrict__ theta,
                                                  float* __restrict__ ax) {
  __shared__ float sTh[HDIM * NRANK];
  __shared__ float sAx[NRANK * HDIM];
  int g = blockIdx.x, tid = threadIdx.x;
  for (int i = tid; i < HDIM * NRANK; i += 256) sTh[i] = theta[i];
  for (int i = tid; i < NRANK * HDIM; i += 256) sAx[i] = 0.f;
  __syncthreads();
  for (int m = tid; m < MNODES; m += 256) {
    const float* row = h + ((size_t)g * MNODES + m) * HDIM;
    float al[NRANK] = {0.f, 0.f, 0.f, 0.f};
    for (int c = 0; c < HDIM; ++c) {
      float x = row[c];
#pragma unroll
      for (int k = 0; k < NRANK; ++k) al[k] += x * sTh[c * NRANK + k];
    }
#pragma unroll
    for (int k = 0; k < NRANK; ++k)
      for (int c = 0; c < HDIM; ++c) atomicAdd(&sAx[k * HDIM + c], al[k] * row[c]);
  }
  __syncthreads();
  for (int i = tid; i < NRANK * HDIM; i += 256) ax[(size_t)g * NRANK * HDIM + i] = sAx[i];
}

// ---------------- mamba over (NGRAPHS, 4, 64); emit last timestep -> xm ----------------
struct MambaArgs {
  const float* ax;
  const float *norm, *in_w, *conv_w, *conv_b, *x_w, *dt_w, *dt_b, *A_log, *Dp, *out_w, *normf;
  f16_t* xm16;
};
__global__ __launch_bounds__(128) void k_mamba(MambaArgs P) {
  __shared__ float sRes[4 * HDIM];
  __shared__ float sHn[4 * HDIM];
  __shared__ float sHsRaw[4 * IDIM];
  __shared__ float sHs[4 * IDIM];
  __shared__ float sSSM[4 * (NRANK + 2 * NSTATE)];   // 4 x 36
  __shared__ float sY[IDIM];
  __shared__ float sO[HDIM];
  const int g = blockIdx.x, tid = threadIdx.x;

  for (int i = tid; i < 4 * HDIM; i += 128) sRes[i] = P.ax[(size_t)g * 4 * HDIM + i];
  __syncthreads();
  if (tid < 4) {                                     // rmsnorm per timestep
    float ss = 0.f;
    for (int c = 0; c < HDIM; ++c) ss += sRes[tid * HDIM + c] * sRes[tid * HDIM + c];
    float rs = rsqrtf(ss * (1.f / HDIM) + EPSV);
    for (int c = 0; c < HDIM; ++c) sHn[tid * HDIM + c] = sRes[tid * HDIM + c] * rs * P.norm[c];
  }
  __syncthreads();
  float gate3;
  {                                                  // in-proj (thread = channel)
    const int i = tid;
    for (int t = 0; t < 4; ++t) {
      float a = 0.f;
      for (int c = 0; c < HDIM; ++c) a += sHn[t * HDIM + c] * P.in_w[c * (2 * IDIM) + i];
      sHsRaw[t * IDIM + i] = a;
    }
    float ga = 0.f;
    for (int c = 0; c < HDIM; ++c) ga += sHn[3 * HDIM + c] * P.in_w[c * (2 * IDIM) + IDIM + i];
    gate3 = ga;
  }
  __syncthreads();
  {                                                  // causal depthwise conv + silu
    const int i = tid;
    for (int t = 0; t < 4; ++t) {
      float a = P.conv_b[i];
#pragma unroll
      for (int k = 0; k < 4; ++k) {
        int ts = t - 3 + k;
        if (ts >= 0) a += sHsRaw[ts * IDIM + i] * P.conv_w[i * 4 + k];
      }
      sHs[t * IDIM + i] = silu_f(a);
    }
  }
  __syncthreads();
  if (tid < NRANK + 2 * NSTATE) {                    // ssm projection (36 outs / t)
    for (int t = 0; t < 4; ++t) {
      float a = 0.f;
      for (int i = 0; i < IDIM; ++i) a += sHs[t * IDIM + i] * P.x_w[i * (NRANK + 2 * NSTATE) + tid];
      sSSM[t * (NRANK + 2 * NSTATE) + tid] = a;
    }
  }
  __syncthreads();
  {                                                  // selective-scan recurrence
    const int i = tid;
    float st[NSTATE];
#pragma unroll
    for (int n = 0; n < NSTATE; ++n) st[n] = 0.f;
    float Ai[NSTATE];
#pragma unroll
    for (int n = 0; n < NSTATE; ++n) Ai[n] = -__expf(P.A_log[i * NSTATE + n]);
    float y3 = 0.f;
    for (int t = 0; t < 4; ++t) {
      const float* ssm = sSSM + t * (NRANK + 2 * NSTATE);
      float dtr = P.dt_b[i];
#pragma unroll
      for (int r = 0; r < NRANK; ++r) dtr += ssm[r] * P.dt_w[r * IDIM + i];
      float dt = (dtr > 20.f) ? dtr : log1pf(__expf(dtr));
      float hs = sHs[t * IDIM + i];
      float yt = 0.f;
#pragma unroll
      for (int n = 0; n < NSTATE; ++n) {
        float dA = __expf(dt * Ai[n]);
        st[n] = dA * st[n] + dt * ssm[NRANK + n] * hs;
        yt += st[n] * ssm[NRANK + NSTATE + n];
      }
      if (t == 3) y3 = yt;
    }
    float y = y3 + sHs[3 * IDIM + i] * P.Dp[i];
    sY[i] = y * silu_f(gate3);
  }
  __syncthreads();
  if (tid < HDIM) {                                  // out-proj + residual
    float o = 0.f;
    for (int i = 0; i < IDIM; ++i) o += sY[i] * P.out_w[i * HDIM + tid];
    sO[tid] = sRes[3 * HDIM + tid] + o;
  }
  __syncthreads();
  if (tid < HDIM) {                                  // final rmsnorm
    float ss = 0.f;
    for (int c = 0; c < HDIM; ++c) ss += sO[c] * sO[c];
    float rs = rsqrtf(ss * (1.f / HDIM) + EPSV);
    P.xm16[(size_t)g * HDIM + tid] = (f16_t)(sO[tid] * rs * P.normf[tid]);
  }
}

// ---------------- merge: concat(xs0,xs1,xs2,xm) 256->64(LN,ReLU)->64 +res +LN -------
struct MergeArgs {
  const f16_t *xs0, *xs1, *xs2, *xm16;
  const int* batch;
  const f16_t *W1t, *W2t;                  // transposed: 64x256, 64x64 (f16)
  const float *b1, *ln1g, *ln1b, *b2;
  const float *lng, *lnb;                  // layer ln
  float* h; f16_t* h16;                    // identity in, new h out (in place)
};
__global__ __launch_bounds__(128) void k_merge(MergeArgs A) {
  __shared__ f16_t sA[16 * 264];
  __shared__ f16_t sW[HDIM * 264];         // 64 x 256 transposed, ldk = 264
  __shared__ float sC[16 * 64];
  const int tid = threadIdx.x, wave = tid >> 5, lane = tid & 31;
  const int n0 = blockIdx.x * 16;
  const int hi = (lane >> 4), col = wave * 16 + (lane & 15);

  for (int i = tid; i < 16 * 256; i += 128) {
    int r = i >> 8, c = i & 255;
    size_t node = (size_t)(n0 + r);
    f16_t v;
    if      (c < 64)  v = A.xs0[node * HDIM + c];
    else if (c < 128) v = A.xs1[node * HDIM + (c - 64)];
    else if (c < 192) v = A.xs2[node * HDIM + (c - 128)];
    else              v = A.xm16[(size_t)A.batch[node] * HDIM + (c - 192)];
    sA[r * 264 + c] = v;
  }
  for (int i = tid; i < HDIM * 256; i += 128) {      // W1t: 64 x 256
    int o = i >> 8, k = i & 255;
    sW[o * 264 + k] = A.W1t[i];
  }
  __syncthreads();

  v8f acc = {};
#pragma unroll
  for (int k0 = 0; k0 < 256; k0 += 32)
    acc = wmma_step(acc, wmma_load_A(sA, 264, k0, lane), wmma_load_Bt(sW, 264, k0, wave * 16, lane));
#pragma unroll
  for (int r = 0; r < 8; ++r) sC[(r + (hi << 3)) * HDIM + col] = acc[r] + A.b1[col];
  __syncthreads();
  if (tid < 16) row_ln_relu_f16(sC, sA, 264, A.ln1g, A.ln1b, tid);
  for (int i = tid; i < HDIM * HDIM; i += 128) {     // W2t: 64 x 64 (reuse sW, ldk=264)
    int o = i >> 6, k = i & 63;
    sW[o * 264 + k] = A.W2t[i];
  }
  __syncthreads();

  acc = (v8f){};
#pragma unroll
  for (int k0 = 0; k0 < HDIM; k0 += 32)
    acc = wmma_step(acc, wmma_load_A(sA, 264, k0, lane), wmma_load_Bt(sW, 264, k0, wave * 16, lane));
#pragma unroll
  for (int r = 0; r < 8; ++r) {
    int row = r + (hi << 3);
    sC[row * HDIM + col] = acc[r] + A.b2[col] + A.h[(size_t)(n0 + row) * HDIM + col];
  }
  __syncthreads();
  if (tid < 16) {                                    // final layer LN -> h, h16
    int row = tid;
    float m = 0.f;
    for (int c = 0; c < HDIM; ++c) m += sC[row * HDIM + c];
    m *= (1.f / HDIM);
    float v = 0.f;
    for (int c = 0; c < HDIM; ++c) { float d = sC[row * HDIM + c] - m; v += d * d; }
    float rs = rsqrtf(v * (1.f / HDIM) + EPSV);
    for (int c = 0; c < HDIM; ++c) {
      float y = (sC[row * HDIM + c] - m) * rs * A.lng[c] + A.lnb[c];
      A.h[(size_t)(n0 + row) * HDIM + c]   = y;
      A.h16[(size_t)(n0 + row) * HDIM + c] = (f16_t)y;
    }
  }
}

// ---------------- pooled = segment_sum(h) per graph ----------------
__global__ __launch_bounds__(256) void k_pool(const float* __restrict__ h, float* __restrict__ pooled) {
  __shared__ float sP[4 * HDIM];
  int g = blockIdx.x, tid = threadIdx.x;
  int c = tid & 63, grp = tid >> 6;
  float s = 0.f;
  for (int m = grp; m < MNODES; m += 4) s += h[((size_t)g * MNODES + m) * HDIM + c];
  sP[grp * HDIM + c] = s;
  __syncthreads();
  if (tid < HDIM) pooled[(size_t)g * HDIM + tid] = sP[tid] + sP[64 + tid] + sP[128 + tid] + sP[192 + tid];
}

// ---------------- decoder MLP (64 rows, tiny) ----------------
struct DecArgs {
  const float* pooled;
  const float *w0, *b0, *g0, *bb0;
  const float *w1, *b1, *g1, *bb1;
  const float *wo, *bo;
  float* out;
};
__global__ __launch_bounds__(64) void k_dec(DecArgs A) {
  int g = threadIdx.x;
  float v[HDIM], t[HDIM];
  for (int c = 0; c < HDIM; ++c) v[c] = A.pooled[(size_t)g * HDIM + c];
  const float* ws[2] = {A.w0, A.w1};
  const float* bs[2] = {A.b0, A.b1};
  const float* gs[2] = {A.g0, A.g1};
  const float* es[2] = {A.bb0, A.bb1};
  for (int l = 0; l < 2; ++l) {
    for (int o = 0; o < HDIM; ++o) {
      float a = bs[l][o];
      for (int c = 0; c < HDIM; ++c) a += v[c] * ws[l][c * HDIM + o];
      t[o] = a;
    }
    float m = 0.f;
    for (int o = 0; o < HDIM; ++o) m += t[o];
    m *= (1.f / HDIM);
    float q = 0.f;
    for (int o = 0; o < HDIM; ++o) { float d = t[o] - m; q += d * d; }
    float rs = rsqrtf(q * (1.f / HDIM) + EPSV);
    for (int o = 0; o < HDIM; ++o) v[o] = fmaxf((t[o] - m) * rs * gs[l][o] + es[l][o], 0.f);
  }
  for (int o = 0; o < OUT_DIM; ++o) {
    float a = A.bo[o];
    for (int c = 0; c < HDIM; ++c) a += v[c] * A.wo[c * OUT_DIM + o];
    A.out[(size_t)g * OUT_DIM + o] = a;
  }
}

// ---------------- host orchestration ----------------
extern "C" void kernel_launch(void* const* d_in, const int* in_sizes, int n_in,
                              void* d_out, int out_size, void* d_ws, size_t ws_size,
                              hipStream_t stream) {
  (void)in_sizes; (void)n_in; (void)out_size; (void)ws_size;
  const float* x_in      = (const float*)d_in[0];
  const int*   edge_idx  = (const int*)d_in[1];
  const int*   batch     = (const int*)d_in[2];
  const float* edge_attr = (const float*)d_in[3];
  const int*   e_src = edge_idx;
  const int*   e_dst = edge_idx + NEDGES;

  // -------- params: jax tree-flatten (sorted dict keys) leaf order --------
  int ci = 4;
  auto P = [&]() { return (const float*)d_in[ci++]; };
  // dec
  const float *dec_b0 = P(), *dec_ln0b = P(), *dec_ln0g = P(), *dec_w0 = P();
  const float *dec_b1 = P(), *dec_ln1b = P(), *dec_ln1g = P(), *dec_w1 = P();
  const float *dec_bo = P(), *dec_wo = P();
  // enc
  const float *enc_b0 = P(), *enc_ln0b = P(), *enc_ln0g = P(), *enc_w0 = P();
  const float *enc_b1 = P(), *enc_ln1b = P(), *enc_ln1g = P(), *enc_w1 = P();
  const float *enc_bo = P(), *enc_wo = P();
  // layers
  struct ConvP { const float *bnd_b,*bnd_w,*fn_b,*fn_g,*b1,*ln1b,*ln1g,*w1,*b2,*w2,*sob_b,*sob_w; };
  struct LayerP {
    ConvP conv[3];
    const float *ln_b,*ln_g;
    const float *A_log,*D,*conv_b,*conv_w,*dt_b,*dt_w,*in_w,*norm,*normf,*out_w,*x_w;
    const float *mg_b1,*mg_ln1b,*mg_ln1g,*mg_w1,*mg_b2,*mg_w2;
    const float *theta;
  };
  LayerP L[3];
  for (int l = 0; l < 3; ++l) {
    for (int c = 0; c < 3; ++c) {
      ConvP& cp = L[l].conv[c];
      cp.bnd_b = P(); cp.bnd_w = P(); cp.fn_b = P(); cp.fn_g = P();
      cp.b1 = P(); cp.ln1b = P(); cp.ln1g = P(); cp.w1 = P();
      cp.b2 = P(); cp.w2 = P(); cp.sob_b = P(); cp.sob_w = P();
    }
    L[l].ln_b = P(); L[l].ln_g = P();
    L[l].A_log = P(); L[l].D = P(); L[l].conv_b = P(); L[l].conv_w = P();
    L[l].dt_b = P(); L[l].dt_w = P(); L[l].in_w = P(); L[l].norm = P();
    L[l].normf = P(); L[l].out_w = P(); L[l].x_w = P();
    L[l].mg_b1 = P(); L[l].mg_ln1b = P(); L[l].mg_ln1g = P(); L[l].mg_w1 = P();
    L[l].mg_b2 = P(); L[l].mg_w2 = P();
    L[l].theta = P();
  }

  // -------- workspace layout --------
  char* wp = (char*)d_ws;
  auto take = [&](size_t bytes) -> void* {
    void* p = (void*)wp;
    wp += (bytes + 255) & ~(size_t)255;
    return p;
  };
  float* h32   = (float*)take((size_t)NNODES * HDIM * 4);
  f16_t* h16   = (f16_t*)take((size_t)NNODES * HDIM * 2);
  float* aggr[3]; f16_t* xs16[3];
  for (int c = 0; c < 3; ++c) aggr[c] = (float*)take((size_t)NNODES * HDIM * 4);
  for (int c = 0; c < 3; ++c) xs16[c] = (f16_t*)take((size_t)NNODES * HDIM * 2);
  float* ax    = (float*)take((size_t)NGRAPHS * NRANK * HDIM * 4);
  f16_t* xm16  = (f16_t*)take((size_t)NGRAPHS * HDIM * 2);
  float* pooled= (float*)take((size_t)NGRAPHS * HDIM * 4);
  f16_t* we0   = (f16_t*)take((size_t)IN_DIM * HDIM * 2);
  f16_t* we1   = (f16_t*)take((size_t)HDIM * HDIM * 2);
  f16_t* we2   = (f16_t*)take((size_t)HDIM * HDIM * 2);
  f16_t* cw1[3]; f16_t* cw2[3];
  for (int c = 0; c < 3; ++c) cw1[c] = (f16_t*)take((size_t)IDIM * HDIM * 2);
  for (int c = 0; c < 3; ++c) cw2[c] = (f16_t*)take((size_t)HDIM * HDIM * 2);
  f16_t* mw1   = (f16_t*)take((size_t)256 * HDIM * 2);
  f16_t* mw2   = (f16_t*)take((size_t)HDIM * HDIM * 2);

  // transpose-cast: src is (K x O) f32, dst is (O x K) f16
  auto castT = [&](const float* s, f16_t* d, int K, int O) {
    int n = K * O;
    k_cast_f16_t<<<(n + 255) / 256, 256, 0, stream>>>(s, d, K, O);
  };

  // -------- encoder --------
  castT(enc_w0, we0, IN_DIM, HDIM);
  castT(enc_w1, we1, HDIM, HDIM);
  castT(enc_wo, we2, HDIM, HDIM);   // mlp out.w is the third GEMM weight
  {
    EncArgs a{x_in, we0, we1, we2, enc_b0, enc_ln0g, enc_ln0b,
              enc_b1, enc_ln1g, enc_ln1b, enc_bo, h32, h16};
    k_encoder<<<NNODES / 16, 128, 0, stream>>>(a);
  }

  // -------- layers --------
  for (int l = 0; l < 3; ++l) {
    for (int c = 0; c < 3; ++c) {
      castT(L[l].conv[c].w1, cw1[c], IDIM, HDIM);
      castT(L[l].conv[c].w2, cw2[c], HDIM, HDIM);
    }
    castT(L[l].mg_w1, mw1, 256, HDIM);
    castT(L[l].mg_w2, mw2, HDIM, HDIM);

    for (int c = 0; c < 3; ++c) {
      (void)hipMemsetAsync(aggr[c], 0, (size_t)NNODES * HDIM * 4, stream);
      ConvArgs a{e_src, e_dst, edge_attr, h16, cw1[c], cw2[c],
                 L[l].conv[c].b1, L[l].conv[c].ln1g, L[l].conv[c].ln1b, L[l].conv[c].b2,
                 L[l].conv[c].sob_w, L[l].conv[c].sob_b, L[l].conv[c].bnd_w, L[l].conv[c].bnd_b,
                 aggr[c]};
      k_edge_conv<<<NEDGES / 16, 128, 0, stream>>>(a);
    }
    for (int c = 0; c < 3; ++c)
      k_fn_ln<<<NNODES / 8, 256, 0, stream>>>(h32, aggr[c], L[l].conv[c].fn_g,
                                              L[l].conv[c].fn_b, xs16[c]);
    k_alpha_ax<<<NGRAPHS, 256, 0, stream>>>(h32, L[l].theta, ax);
    {
      MambaArgs a{ax, L[l].norm, L[l].in_w, L[l].conv_w, L[l].conv_b, L[l].x_w,
                  L[l].dt_w, L[l].dt_b, L[l].A_log, L[l].D, L[l].out_w, L[l].normf, xm16};
      k_mamba<<<NGRAPHS, 128, 0, stream>>>(a);
    }
    {
      MergeArgs a{xs16[0], xs16[1], xs16[2], xm16, batch, mw1, mw2,
                  L[l].mg_b1, L[l].mg_ln1g, L[l].mg_ln1b, L[l].mg_b2,
                  L[l].ln_g, L[l].ln_b, h32, h16};
      k_merge<<<NNODES / 16, 128, 0, stream>>>(a);
    }
  }

  // -------- pool + decode --------
  k_pool<<<NGRAPHS, 256, 0, stream>>>(h32, pooled);
  {
    DecArgs a{pooled, dec_w0, dec_b0, dec_ln0g, dec_ln0b,
              dec_w1, dec_b1, dec_ln1g, dec_ln1b, dec_wo, dec_bo, (float*)d_out};
    k_dec<<<1, 64, 0, stream>>>(a);
  }
}
